// MoE_18614388261659
// MI455X (gfx1250) — matrix operand order repaired and checked by reading, not compiled, not run on hardware
//
#include <hip/hip_runtime.h>
#include <hip/hip_bf16.h>

// Problem dims (fixed by reference setup_inputs)
#define B_DIM   2
#define N_DIM   4096
#define D_DIM   512
#define E_DIM   16
#define H_DIM   2048
#define CAP     1024
#define TOKENS  (B_DIM * N_DIM)            // 8192
#define SLOTS_E (B_DIM * CAP)              // 2048 slots per expert
#define NBLK    (TOKENS / 256)             // 32 gating blocks (16 per batch)
#define MTILE   64                         // slot rows per workgroup

typedef __attribute__((ext_vector_type(16))) __bf16 bf16x16;
typedef __attribute__((ext_vector_type(8)))  float  floatx8;

// ---------------------------------------------------------------------------
// G1: gating — logits = x @ w_gating, softmax, top1/top2, normalized gates,
//     and per-block pre-capacity histograms for top1/top2 experts.
// ---------------------------------------------------------------------------
__global__ __launch_bounds__(256)
void moe_gate_kernel(const float* __restrict__ x, const float* __restrict__ wg,
                     int* __restrict__ idx1, int* __restrict__ idx2,
                     float* __restrict__ g1, float* __restrict__ g2,
                     int* __restrict__ blockcnt1, int* __restrict__ blockcnt2) {
    __shared__ float wgs[D_DIM * E_DIM];
    __shared__ int c1[E_DIM], c2[E_DIM];

    const int tid = threadIdx.x;
    for (int i = tid; i < D_DIM * E_DIM; i += 256) wgs[i] = wg[i];
    if (tid < E_DIM) { c1[tid] = 0; c2[tid] = 0; }
    __syncthreads();

    const int t = blockIdx.x * 256 + tid;
    const float* xr = x + (size_t)t * D_DIM;

    float acc[E_DIM];
#pragma unroll
    for (int e = 0; e < E_DIM; ++e) acc[e] = 0.f;

    for (int k = 0; k < D_DIM; k += 4) {
        float4 xv = *reinterpret_cast<const float4*>(xr + k);
#pragma unroll
        for (int e = 0; e < E_DIM; ++e) {
            acc[e] += xv.x * wgs[(k + 0) * E_DIM + e] + xv.y * wgs[(k + 1) * E_DIM + e]
                    + xv.z * wgs[(k + 2) * E_DIM + e] + xv.w * wgs[(k + 3) * E_DIM + e];
        }
    }

    float mx = acc[0];
#pragma unroll
    for (int e = 1; e < E_DIM; ++e) mx = fmaxf(mx, acc[e]);
    float p[E_DIM]; float s = 0.f;
#pragma unroll
    for (int e = 0; e < E_DIM; ++e) { p[e] = __expf(acc[e] - mx); s += p[e]; }
    const float inv = 1.f / s;

    int i1 = 0; float r1 = p[0];
#pragma unroll
    for (int e = 1; e < E_DIM; ++e) if (p[e] > r1) { r1 = p[e]; i1 = e; }
    int i2 = -1; float r2 = -1.f;
#pragma unroll
    for (int e = 0; e < E_DIM; ++e) if (e != i1 && p[e] > r2) { r2 = p[e]; i2 = e; }

    r1 *= inv; r2 *= inv;
    const float denom = r1 + r2 + 1e-9f;
    idx1[t] = i1; idx2[t] = i2;
    g1[t] = r1 / denom; g2[t] = r2 / denom;

    atomicAdd(&c1[i1], 1);
    atomicAdd(&c2[i2], 1);
    __syncthreads();
    if (tid < E_DIM) {
        blockcnt1[blockIdx.x * E_DIM + tid] = c1[tid];
        blockcnt2[blockIdx.x * E_DIM + tid] = c2[tid];
    }
}

// ---------------------------------------------------------------------------
// G2: per-(batch,expert) exclusive block prefix + mask_1_count = min(total,cap)
// ---------------------------------------------------------------------------
__global__ void moe_scan_kernel(const int* __restrict__ blockcnt1,
                                const int* __restrict__ blockcnt2,
                                int* __restrict__ blockoff1,
                                int* __restrict__ blockoff2,
                                int* __restrict__ m1cnt) {
    const int tid = threadIdx.x;
    if (tid >= B_DIM * E_DIM) return;
    const int b = tid >> 4, e = tid & 15;
    int off = 0;
    for (int blk = 0; blk < NBLK / B_DIM; ++blk) {
        const int g = b * (NBLK / B_DIM) + blk;
        blockoff1[g * E_DIM + e] = off;
        off += blockcnt1[g * E_DIM + e];
    }
    m1cnt[b * E_DIM + e] = off < CAP ? off : CAP;
    off = 0;
    for (int blk = 0; blk < NBLK / B_DIM; ++blk) {
        const int g = b * (NBLK / B_DIM) + blk;
        blockoff2[g * E_DIM + e] = off;
        off += blockcnt2[g * E_DIM + e];
    }
}

// ---------------------------------------------------------------------------
// G3: token-order ranks (wave32 ballot + cross-wave LDS prefix), capacity
//     dropping, slot map construction.
// ---------------------------------------------------------------------------
__global__ __launch_bounds__(256)
void moe_assign_kernel(const int* __restrict__ idx1, const int* __restrict__ idx2,
                       const float* __restrict__ g1, const float* __restrict__ g2,
                       const int* __restrict__ blockoff1, const int* __restrict__ blockoff2,
                       const int* __restrict__ m1cnt,
                       int* __restrict__ slot_token, float* __restrict__ slot_gate) {
    __shared__ int wc1[8][E_DIM], wc2[8][E_DIM];

    const int tid = threadIdx.x, lane = tid & 31, wave = tid >> 5;
    const int t = blockIdx.x * 256 + tid;
    const int b = t >> 12;                 // token / 4096
    const int e1 = idx1[t], e2 = idx2[t];
    const unsigned lt = (1u << lane) - 1u;

    int rank1 = 0, rank2 = 0;
#pragma unroll
    for (int e = 0; e < E_DIM; ++e) {
        unsigned b1 = __builtin_amdgcn_ballot_w32(e1 == e);
        unsigned b2 = __builtin_amdgcn_ballot_w32(e2 == e);
        if (e == e1) rank1 = __popc(b1 & lt);
        if (e == e2) rank2 = __popc(b2 & lt);
        if (lane == e) { wc1[wave][e] = __popc(b1); wc2[wave][e] = __popc(b2); }
    }
    __syncthreads();

    int pos1 = blockoff1[blockIdx.x * E_DIM + e1] + rank1;
    int pos2 = blockoff2[blockIdx.x * E_DIM + e2] + rank2 + m1cnt[b * E_DIM + e2];
#pragma unroll
    for (int w = 0; w < 8; ++w) {
        if (w < wave) {
            pos1 += wc1[w][e1];
            pos2 += wc2[w][e2];
        }
    }

    if (pos1 < CAP) {
        const int s = e1 * SLOTS_E + b * CAP + pos1;
        slot_token[s] = t; slot_gate[s] = g1[t];
    }
    if (pos2 < CAP) {
        const int s = e2 * SLOTS_E + b * CAP + pos2;
        slot_token[s] = t; slot_gate[s] = g2[t];
    }
}

// ---------------------------------------------------------------------------
// Weight transpose + fp32->bf16 convert:  src fp32 [e][K][N] -> dst bf16 [e][N][K]
// 64x64 tiles through padded LDS; coalesced reads and packed-b32 writes.
// ---------------------------------------------------------------------------
__global__ __launch_bounds__(256)
void transpose_cvt_kernel(const float* __restrict__ src, __bf16* __restrict__ dst,
                          int K, int N) {
    __shared__ __bf16 tileT[64 * 66];          // [n][k], stride 66 avoids conflicts
    const int e  = blockIdx.z;
    const int n0 = blockIdx.x * 64;
    const int k0 = blockIdx.y * 64;
    const float* s = src + (size_t)e * K * N;
    __bf16*      d = dst + (size_t)e * K * N;
    const int tid = threadIdx.x;

    for (int i = tid; i < 64 * 64; i += 256) {
        const int r = i >> 6, c = i & 63;       // r: k within tile, c: n within tile
        tileT[c * 66 + r] = (__bf16)s[(size_t)(k0 + r) * N + n0 + c];
    }
    __syncthreads();

    unsigned* du = reinterpret_cast<unsigned*>(d);
    for (int i = tid; i < 64 * 32; i += 256) {
        const int r = i >> 5, cu = i & 31;      // r: n within tile, cu: k-pair
        const unsigned pk = *reinterpret_cast<const unsigned*>(&tileT[r * 66 + 2 * cu]);
        du[((size_t)(n0 + r) * K + k0) / 2 + cu] = pk;
    }
}

// ---------------------------------------------------------------------------
// Fused expert FFN: per (expert, 64-slot tile):
//   H = relu(Xtile @ W1)  (bf16 WMMA, H kept in 256KB of LDS)
//   O = H @ W2            (bf16 WMMA)
//   out[token] += gate * O  via global_atomic_add_f32
// PRE=true: B-fragments are 2x global_load_b128 from pre-transposed bf16 weights.
// PRE=false: fallback, direct strided fp32 loads + cvt.
// ---------------------------------------------------------------------------
template <bool PRE>
__global__ __launch_bounds__(256)
void moe_ffn_kernel(const float* __restrict__ x,
                    const float* __restrict__ w1, const float* __restrict__ w2,
                    const __bf16* __restrict__ w1t, const __bf16* __restrict__ w2t,
                    const int* __restrict__ slot_token,
                    const float* __restrict__ slot_gate,
                    float* __restrict__ out) {
    __shared__ __bf16 Xs[MTILE * D_DIM];       //  64 KB
    __shared__ __bf16 Hs[MTILE * H_DIM];       // 256 KB  -> 320 KB total (CDNA5 WGP LDS)

    const int e     = blockIdx.y;
    const int tile  = blockIdx.x;              // 0..31 (64 rows each)
    const int sBase = e * SLOTS_E + tile * MTILE;

    const int tid  = threadIdx.x;
    const int lane = tid & 31;
    const int wave = tid >> 5;
    const int half = lane >> 4;                // 0: lanes 0-15, 1: lanes 16-31
    const int nc   = lane & 15;                // column within 16-wide tile
    const int m    = lane & 15;                // row within 16-tall A tile

    // ---- gather X tile (fp32 -> bf16), empty slots = zero rows ----
    const float4* x4 = reinterpret_cast<const float4*>(x);
    for (int flat = tid; flat < MTILE * (D_DIM / 4); flat += 256) {
        const int row = flat >> 7;             // D_DIM/4 == 128
        const int c4  = flat & 127;
        const int tok = slot_token[sBase + row];
        float4 v = make_float4(0.f, 0.f, 0.f, 0.f);
        if (tok >= 0) v = x4[(size_t)tok * (D_DIM / 4) + c4];
        __bf16* dst = &Xs[row * D_DIM + c4 * 4];
        dst[0] = (__bf16)v.x; dst[1] = (__bf16)v.y;
        dst[2] = (__bf16)v.z; dst[3] = (__bf16)v.w;
    }
    __syncthreads();

    union AF { bf16x16 v; uint4 q[2]; };
    union BU { bf16x16 v; uint4 q[2]; __bf16 h[16]; };

    // ---- stage 1: H = relu(X @ W1) ----
    const float*  w1e  = w1  + (size_t)e * D_DIM * H_DIM;
    const __bf16* w1te = PRE ? (w1t + (size_t)e * D_DIM * H_DIM) : nullptr;
    for (int nt = wave; nt < H_DIM / 16; nt += 8) {
        const int n0 = nt * 16;
        floatx8 acc[4];
#pragma unroll
        for (int mt = 0; mt < 4; ++mt) acc[mt] = (floatx8){0.f,0.f,0.f,0.f,0.f,0.f,0.f,0.f};
#pragma unroll 2
        for (int ks = 0; ks < D_DIM / 32; ++ks) {
            const int k0 = ks * 32;
            BU Bf;
            if (PRE) {
                // lane holds column n0+nc; K = k0 + half*16 + i  (contiguous bf16)
                const __bf16* bp = w1te + (size_t)(n0 + nc) * D_DIM + k0 + half * 16;
                __builtin_prefetch(bp + 64, 0, 3);
                Bf.q[0] = *reinterpret_cast<const uint4*>(bp);
                Bf.q[1] = *reinterpret_cast<const uint4*>(bp + 8);
            } else {
                const float* bp = w1e + (size_t)(k0 + half * 16) * H_DIM + n0 + nc;
#pragma unroll
                for (int i = 0; i < 16; ++i) Bf.h[i] = (__bf16)bp[(size_t)i * H_DIM];
            }
            AF a[4];
#pragma unroll
            for (int mt = 0; mt < 4; ++mt) {
                const __bf16* ap = &Xs[(16 * mt + m) * D_DIM + k0 + half * 16];
                a[mt].q[0] = *reinterpret_cast<const uint4*>(ap);       // K 0..7  (+half*16)
                a[mt].q[1] = *reinterpret_cast<const uint4*>(ap + 8);   // K 8..15 (+half*16)
            }
#pragma unroll
            for (int mt = 0; mt < 4; ++mt)
                acc[mt] = __builtin_amdgcn_wmma_f32_16x16x32_bf16(false, a[mt].v, false, Bf.v,
                                                                  (short)0, acc[mt], false, false);
        }
#pragma unroll
        for (int mt = 0; mt < 4; ++mt) {
#pragma unroll
            for (int r = 0; r < 8; ++r) {
                float v = acc[mt][r]; v = v > 0.f ? v : 0.f;            // relu
                Hs[(16 * mt + 8 * half + r) * H_DIM + n0 + nc] = (__bf16)v;
            }
        }
    }
    __syncthreads();

    // ---- stage 2: O = H @ W2, scatter-add weighted by gate ----
    const float*  w2e  = w2  + (size_t)e * H_DIM * D_DIM;
    const __bf16* w2te = PRE ? (w2t + (size_t)e * H_DIM * D_DIM) : nullptr;
    for (int nt = wave; nt < D_DIM / 16; nt += 8) {
        const int n0 = nt * 16;
        floatx8 acc[4];
#pragma unroll
        for (int mt = 0; mt < 4; ++mt) acc[mt] = (floatx8){0.f,0.f,0.f,0.f,0.f,0.f,0.f,0.f};
#pragma unroll 2
        for (int ks = 0; ks < H_DIM / 32; ++ks) {
            const int k0 = ks * 32;
            BU Bf;
            if (PRE) {
                const __bf16* bp = w2te + (size_t)(n0 + nc) * H_DIM + k0 + half * 16;
                __builtin_prefetch(bp + 64, 0, 3);
                Bf.q[0] = *reinterpret_cast<const uint4*>(bp);
                Bf.q[1] = *reinterpret_cast<const uint4*>(bp + 8);
            } else {
                const float* bp = w2e + (size_t)(k0 + half * 16) * D_DIM + n0 + nc;
#pragma unroll
                for (int i = 0; i < 16; ++i) Bf.h[i] = (__bf16)bp[(size_t)i * D_DIM];
            }
            AF a[4];
#pragma unroll
            for (int mt = 0; mt < 4; ++mt) {
                const __bf16* ap = &Hs[(16 * mt + m) * H_DIM + k0 + half * 16];
                a[mt].q[0] = *reinterpret_cast<const uint4*>(ap);
                a[mt].q[1] = *reinterpret_cast<const uint4*>(ap + 8);
            }
#pragma unroll
            for (int mt = 0; mt < 4; ++mt)
                acc[mt] = __builtin_amdgcn_wmma_f32_16x16x32_bf16(false, a[mt].v, false, Bf.v,
                                                                  (short)0, acc[mt], false, false);
        }
#pragma unroll
        for (int mt = 0; mt < 4; ++mt) {
#pragma unroll
            for (int r = 0; r < 8; ++r) {
                const int lr = 16 * mt + 8 * half + r;
                const int tk = slot_token[sBase + lr];
                if (tk >= 0) {
                    const float g = slot_gate[sBase + lr];
                    atomicAdd(&out[(size_t)tk * D_DIM + n0 + nc], g * acc[mt][r]);
                }
            }
        }
    }
}

// ---------------------------------------------------------------------------
extern "C" void kernel_launch(void* const* d_in, const int* in_sizes, int n_in,
                              void* d_out, int out_size, void* d_ws, size_t ws_size,
                              hipStream_t stream) {
    const float* x  = (const float*)d_in[0];
    const float* wg = (const float*)d_in[1];
    const float* w1 = (const float*)d_in[2];
    const float* w2 = (const float*)d_in[3];
    float* out = (float*)d_out;

    char* ws = (char*)d_ws;
    int*   idx1 = (int*)(ws + 0);                       // 32768 B
    int*   idx2 = (int*)(ws + 32768);                   // 32768 B
    float* g1   = (float*)(ws + 65536);                 // 32768 B
    float* g2   = (float*)(ws + 98304);                 // 32768 B
    int*   bc1  = (int*)(ws + 131072);                  //  2048 B
    int*   bc2  = (int*)(ws + 133120);                  //  2048 B
    int*   bo1  = (int*)(ws + 135168);                  //  2048 B
    int*   bo2  = (int*)(ws + 137216);                  //  2048 B
    int*   m1c  = (int*)(ws + 139264);                  //   128 B
    int*   slot_token = (int*)(ws + 139392);            // 131072 B
    float* slot_gate  = (float*)(ws + 270464);          // 131072 B

    const size_t WSZ   = (size_t)E_DIM * D_DIM * H_DIM * sizeof(__bf16);  // 33.5 MB
    const size_t OFF1  = 524288;
    const size_t OFF2  = OFF1 + WSZ;
    const size_t NEED  = OFF2 + WSZ;                    // ~64.5 MB
    __bf16* w1t = (__bf16*)(ws + OFF1);
    __bf16* w2t = (__bf16*)(ws + OFF2);
    const bool pre = (ws_size >= NEED);

    hipMemsetAsync(out, 0, (size_t)TOKENS * D_DIM * sizeof(float), stream);
    hipMemsetAsync(slot_token, 0xFF, (size_t)E_DIM * SLOTS_E * sizeof(int), stream);

    if (pre) {
        // W1: [e][D][H] -> [e][H][D];  grid = (N/64, K/64, E)
        dim3 gt1(H_DIM / 64, D_DIM / 64, E_DIM);
        transpose_cvt_kernel<<<gt1, 256, 0, stream>>>(w1, w1t, D_DIM, H_DIM);
        // W2: [e][H][D] -> [e][D][H]
        dim3 gt2(D_DIM / 64, H_DIM / 64, E_DIM);
        transpose_cvt_kernel<<<gt2, 256, 0, stream>>>(w2, w2t, H_DIM, D_DIM);
    }

    moe_gate_kernel<<<NBLK, 256, 0, stream>>>(x, wg, idx1, idx2, g1, g2, bc1, bc2);
    moe_scan_kernel<<<1, 32, 0, stream>>>(bc1, bc2, bo1, bo2, m1c);
    moe_assign_kernel<<<NBLK, 256, 0, stream>>>(idx1, idx2, g1, g2, bo1, bo2, m1c,
                                                slot_token, slot_gate);

    dim3 grid(SLOTS_E / MTILE, E_DIM);   // (32 tiles, 16 experts)
    if (pre)
        moe_ffn_kernel<true><<<grid, 256, 0, stream>>>(x, w1, w2, w1t, w2t,
                                                       slot_token, slot_gate, out);
    else
        moe_ffn_kernel<false><<<grid, 256, 0, stream>>>(x, w1, w2, nullptr, nullptr,
                                                        slot_token, slot_gate, out);
}
